// Decoder_Attention_37477884625683
// MI455X (gfx1250) — matrix operand order repaired and checked by reading, not compiled, not run on hardware
//
#include <hip/hip_runtime.h>

typedef __attribute__((ext_vector_type(16))) _Float16 v16h;
typedef __attribute__((ext_vector_type(4)))  _Float16 v4h;
typedef __attribute__((ext_vector_type(8)))  float    v8f;

#define WMMA_F16(A, Bm, Cm) __builtin_amdgcn_wmma_f32_16x16x32_f16( \
    false, (A), false, (Bm), (short)0, (Cm), false, false)

static constexpr int B_ = 4, N_ = 1024, C_ = 768, H_ = 12, D_ = 64;
static constexpr int M_ = B_ * N_;   // 4096 token rows over all batches
static constexpr int C3 = 3 * C_;    // 2304

// ---------------------------------------------------------------------------
// f32 -> f16 conversion, 4 elements/thread (b128 load, b64 store)
// ---------------------------------------------------------------------------
__global__ void cvt_f32_f16(const float* __restrict__ in,
                            _Float16* __restrict__ out, int n4) {
  int i = blockIdx.x * 256 + threadIdx.x;
  if (i < n4) {
    float4 f = ((const float4*)in)[i];
    v4h h;
    h[0] = (_Float16)f.x; h[1] = (_Float16)f.y;
    h[2] = (_Float16)f.z; h[3] = (_Float16)f.w;
    ((v4h*)out)[i] = h;
  }
}

// ---------------------------------------------------------------------------
// QKV projection: qkv = x @ qkv_w^T + qkv_b
// One wave -> 16x64 output tile (A tile reused across 4 WMMA columns).
// q,k written [B,H,N,D]; v written transposed [B,H,D,N].
// ---------------------------------------------------------------------------
__global__ __launch_bounds__(32)
void qkv_gemm(const _Float16* __restrict__ x16, const _Float16* __restrict__ w16,
              const float* __restrict__ qkv_b,
              _Float16* __restrict__ qbuf, _Float16* __restrict__ kbuf,
              _Float16* __restrict__ vT) {
  const int lane = threadIdx.x;
  const int l15 = lane & 15, g = lane >> 4;
  const int mtile = blockIdx.y;   // 0..255
  const int nt64  = blockIdx.x;   // 0..35
  const int arow  = mtile * 16 + l15;
  const _Float16* xrow = x16 + (size_t)arow * C_;

  v8f acc[4] = {};
  for (int k0 = 0; k0 < C_; k0 += 32) {
    v16h a;
#pragma unroll
    for (int j = 0; j < 8; ++j) {
      a[j]     = xrow[k0 + 8 * g + j];
      a[j + 8] = xrow[k0 + 16 + 8 * g + j];
    }
#pragma unroll
    for (int t = 0; t < 4; ++t) {
      const int col = nt64 * 64 + t * 16 + l15;
      const _Float16* wr = w16 + (size_t)col * C_ + k0 + 8 * g;
      __builtin_prefetch((const void*)(wr + 32), 0, 3);  // next K-step
      v16h bm;
#pragma unroll
      for (int j = 0; j < 16; ++j) bm[j] = wr[(j & 7) + ((j & 8) << 1)];
      acc[t] = WMMA_F16(a, bm, acc[t]);
    }
  }

  const int b = (mtile * 16) / N_;            // uniform within tile
  const int tokBase = (mtile * 16) % N_ + 8 * g;
#pragma unroll
  for (int t = 0; t < 4; ++t) {
    const int col = nt64 * 64 + t * 16 + l15;
    const float bias = qkv_b[col];
    if (col < C_) {                                    // Q
      const int h = col / D_, d = col % D_;
      _Float16* qp = qbuf + ((size_t)(b * H_ + h) * N_) * D_ + d;
#pragma unroll
      for (int i = 0; i < 8; ++i)
        qp[(size_t)(tokBase + i) * D_] = (_Float16)(acc[t][i] + bias);
    } else if (col < 2 * C_) {                         // K
      const int c = col - C_, h = c / D_, d = c % D_;
      _Float16* kp = kbuf + ((size_t)(b * H_ + h) * N_) * D_ + d;
#pragma unroll
      for (int i = 0; i < 8; ++i)
        kp[(size_t)(tokBase + i) * D_] = (_Float16)(acc[t][i] + bias);
    } else {                                           // V (transposed)
      const int c = col - 2 * C_, h = c / D_, d = c % D_;
      _Float16* vp = vT + ((size_t)(b * H_ + h) * D_ + d) * N_;
#pragma unroll
      for (int i = 0; i < 8; ++i)
        vp[tokBase + i] = (_Float16)(acc[t][i] + bias);
    }
  }
}

// ---------------------------------------------------------------------------
// Fused attention per (b, h, 16-row tile), one wave + 64 KB LDS:
//   scores (WMMA) -> mask -> softmax -> attn out (f32, b128 stores)
//   -> P@V (WMMA) -> ctx f16 [B,N,C]
// ---------------------------------------------------------------------------
__global__ __launch_bounds__(32)
void attn_fused(const _Float16* __restrict__ qbuf, const _Float16* __restrict__ kbuf,
                const _Float16* __restrict__ vT, const int* __restrict__ mask,
                float* __restrict__ attn_out, _Float16* __restrict__ ctx) {
  __shared__ float sc[16 * N_];   // 64 KB of the 320 KB WGP LDS pool
  const int lane = threadIdx.x, l15 = lane & 15, g = lane >> 4;
  const int mtile = blockIdx.x;   // 0..63
  const int h = blockIdx.y, b = blockIdx.z;
  const _Float16* qf = qbuf + (size_t)(b * H_ + h) * N_ * D_;
  const _Float16* kf = kbuf + (size_t)(b * H_ + h) * N_ * D_;
  const _Float16* vf = vT   + (size_t)(b * H_ + h) * D_ * N_;

  // ---- Phase A: 16x1024 score tile, K=D=64 in two WMMA steps ----
  const int arow = mtile * 16 + l15;
  v16h a0, a1;
#pragma unroll
  for (int j = 0; j < 8; ++j) {
    a0[j]     = qf[(size_t)arow * D_ + 0  + 8 * g + j];
    a0[j + 8] = qf[(size_t)arow * D_ + 16 + 8 * g + j];
    a1[j]     = qf[(size_t)arow * D_ + 32 + 8 * g + j];
    a1[j + 8] = qf[(size_t)arow * D_ + 48 + 8 * g + j];
  }
  for (int nt = 0; nt < 64; ++nt) {
    const int colTok = nt * 16 + l15;
    const _Float16* kp = kf + (size_t)colTok * D_ + 8 * g;
    __builtin_prefetch((const void*)(kp + 16 * D_), 0, 3);  // next col tile
    v16h b0, b1;
#pragma unroll
    for (int j = 0; j < 16; ++j) {
      b0[j] = kp[(j & 7) + ((j & 8) << 1)];
      b1[j] = kp[32 + (j & 7) + ((j & 8) << 1)];
    }
    v8f acc = {};
    acc = WMMA_F16(a0, b0, acc);
    acc = WMMA_F16(a1, b1, acc);
#pragma unroll
    for (int i = 0; i < 8; ++i)
      sc[(8 * g + i) * N_ + colTok] = acc[i] * 0.125f;   // scale = D^-0.5
  }
  __syncthreads();

  // ---- Phase B: mask + row softmax; lane = (row l15, half g); float4 ----
  {
    const int r = l15;
    float4* srow4 = (float4*)(sc + r * N_ + g * 512);
    const int4* mrow4 =
        (const int4*)(mask + ((size_t)b * N_ + mtile * 16 + r) * N_ + g * 512);
    const float ninf = -__builtin_inff();
    float mx = ninf;
    for (int c = 0; c < 128; ++c) {
      int4 m = mrow4[c];
      float4 s = srow4[c];
      s.x = m.x ? s.x : ninf;
      s.y = m.y ? s.y : ninf;
      s.z = m.z ? s.z : ninf;
      s.w = m.w ? s.w : ninf;
      srow4[c] = s;
      mx = fmaxf(mx, fmaxf(fmaxf(s.x, s.y), fmaxf(s.z, s.w)));
    }
    mx = fmaxf(mx, __shfl_xor(mx, 16, 32));
    float sum = 0.f;
    for (int c = 0; c < 128; ++c) {
      float4 s = srow4[c];
      s.x = __expf(s.x - mx);
      s.y = __expf(s.y - mx);
      s.z = __expf(s.z - mx);
      s.w = __expf(s.w - mx);
      srow4[c] = s;
      sum += (s.x + s.y) + (s.z + s.w);
    }
    sum += __shfl_xor(sum, 16, 32);
    const float inv = 1.0f / sum;
    float4* aout4 = (float4*)(attn_out +
        ((size_t)(b * H_ + h) * N_ + mtile * 16 + r) * N_ + g * 512);
    for (int c = 0; c < 128; ++c) {
      float4 s = srow4[c];
      s.x *= inv; s.y *= inv; s.z *= inv; s.w *= inv;
      srow4[c] = s;
      aout4[c] = s;   // 128-bit stores: dominant HBM stream
    }
  }
  __syncthreads();

  // ---- Phase C: out(16x64) = P(16x1024) @ V(1024x64), 4 accumulators ----
  v8f acc[4] = {};
  for (int k0 = 0; k0 < N_; k0 += 32) {
    v16h a;
#pragma unroll
    for (int j = 0; j < 8; ++j) {
      a[j]     = (_Float16)sc[l15 * N_ + k0 + 8 * g + j];
      a[j + 8] = (_Float16)sc[l15 * N_ + k0 + 16 + 8 * g + j];
    }
#pragma unroll
    for (int t = 0; t < 4; ++t) {
      const _Float16* vp = vf + (size_t)(t * 16 + l15) * N_ + k0 + 8 * g;
      __builtin_prefetch((const void*)(vp + 32), 0, 3);
      v16h bm;
#pragma unroll
      for (int j = 0; j < 16; ++j) bm[j] = vp[(j & 7) + ((j & 8) << 1)];
      acc[t] = WMMA_F16(a, bm, acc[t]);
    }
  }
  const int tokBase = mtile * 16 + 8 * g;
#pragma unroll
  for (int t = 0; t < 4; ++t) {
    const int d = t * 16 + l15;
#pragma unroll
    for (int i = 0; i < 8; ++i)
      ctx[(size_t)(b * N_ + tokBase + i) * C_ + h * D_ + d] =
          (_Float16)acc[t][i];
  }
}

// ---------------------------------------------------------------------------
// Output projection: out = ctx @ proj_w^T + proj_b   (fp32 out)
// ---------------------------------------------------------------------------
__global__ __launch_bounds__(32)
void proj_gemm(const _Float16* __restrict__ ctx, const _Float16* __restrict__ pw16,
               const float* __restrict__ proj_b, float* __restrict__ out) {
  const int lane = threadIdx.x;
  const int l15 = lane & 15, g = lane >> 4;
  const int mtile = blockIdx.y;   // 0..255
  const int nt64  = blockIdx.x;   // 0..11
  const int arow  = mtile * 16 + l15;
  const _Float16* crow = ctx + (size_t)arow * C_;

  v8f acc[4] = {};
  for (int k0 = 0; k0 < C_; k0 += 32) {
    v16h a;
#pragma unroll
    for (int j = 0; j < 8; ++j) {
      a[j]     = crow[k0 + 8 * g + j];
      a[j + 8] = crow[k0 + 16 + 8 * g + j];
    }
#pragma unroll
    for (int t = 0; t < 4; ++t) {
      const int col = nt64 * 64 + t * 16 + l15;
      const _Float16* wr = pw16 + (size_t)col * C_ + k0 + 8 * g;
      __builtin_prefetch((const void*)(wr + 32), 0, 3);
      v16h bm;
#pragma unroll
      for (int j = 0; j < 16; ++j) bm[j] = wr[(j & 7) + ((j & 8) << 1)];
      acc[t] = WMMA_F16(a, bm, acc[t]);
    }
  }
  const int rowBase = mtile * 16 + 8 * g;
#pragma unroll
  for (int t = 0; t < 4; ++t) {
    const int col = nt64 * 64 + t * 16 + l15;
    const float bias = proj_b[col];
#pragma unroll
    for (int i = 0; i < 8; ++i)
      out[(size_t)(rowBase + i) * C_ + col] = acc[t][i] + bias;
  }
}

// ---------------------------------------------------------------------------
extern "C" void kernel_launch(void* const* d_in, const int* in_sizes, int n_in,
                              void* d_out, int out_size, void* d_ws, size_t ws_size,
                              hipStream_t stream) {
  const float* x      = (const float*)d_in[0];
  const float* qkv_w  = (const float*)d_in[1];
  const float* qkv_b  = (const float*)d_in[2];
  const float* proj_w = (const float*)d_in[3];
  const float* proj_b = (const float*)d_in[4];
  const int*   mask   = (const int*)d_in[5];

  float* out  = (float*)d_out;                 // [B,N,C]
  float* attn = out + (size_t)M_ * C_;         // [B,H,N,N]

  // Workspace carve-up (f16 halves): ~36 MB total
  _Float16* qbuf = (_Float16*)d_ws;                       // 3,145,728
  _Float16* kbuf = qbuf + (size_t)M_ * C_;                // 3,145,728
  _Float16* vT   = kbuf + (size_t)M_ * C_;                // 3,145,728
  _Float16* ctx  = vT   + (size_t)M_ * C_;                // 3,145,728
  _Float16* x16  = ctx  + (size_t)M_ * C_;                // 3,145,728
  _Float16* w16  = x16  + (size_t)M_ * C_;                // 1,769,472
  _Float16* pw16 = w16  + (size_t)C3 * C_;                //   589,824

  const int nX4 = M_ * C_ / 4, nW4 = C3 * C_ / 4, nP4 = C_ * C_ / 4;
  cvt_f32_f16<<<(nX4 + 255) / 256, 256, 0, stream>>>(x, x16, nX4);
  cvt_f32_f16<<<(nW4 + 255) / 256, 256, 0, stream>>>(qkv_w, w16, nW4);
  cvt_f32_f16<<<(nP4 + 255) / 256, 256, 0, stream>>>(proj_w, pw16, nP4);

  qkv_gemm<<<dim3(C3 / 64, M_ / 16), 32, 0, stream>>>(x16, w16, qkv_b,
                                                      qbuf, kbuf, vT);
  attn_fused<<<dim3(N_ / 16, H_, B_), 32, 0, stream>>>(qbuf, kbuf, vT, mask,
                                                       attn, ctx);
  proj_gemm<<<dim3(C_ / 64, M_ / 16), 32, 0, stream>>>(ctx, pw16, proj_b, out);
}
